// KDNet_28089086116383
// MI455X (gfx1250) — matrix operand order repaired and checked by reading, not compiled, not run on hardware
//
#include <hip/hip_runtime.h>
#include <math.h>

typedef __attribute__((ext_vector_type(16))) _Float16 v16h;
typedef __attribute__((ext_vector_type(8)))  float    v8f;
typedef __attribute__((ext_vector_type(4)))  unsigned int v4u;
typedef __attribute__((ext_vector_type(8)))  int      v8i;
typedef __attribute__((ext_vector_type(4)))  int      v4i;

#define B_BATCH 128

#if defined(__HIP_DEVICE_COMPILE__) && __has_builtin(__builtin_amdgcn_tensor_load_to_lds)
#define USE_TDM 1
#else
#define USE_TDM 0
#endif

union FragU { v16h h; unsigned int u[8]; };

__global__ void zero_f32(float* __restrict__ p, int n) {
  int i = blockIdx.x * 256 + threadIdx.x;
  if (i < n) p[i] = 0.0f;
}

#if USE_TDM
// Build D# group0: count=1, lds byte addr, 57-bit global byte addr, type=2.
__device__ __forceinline__ v4u tdm_g0(unsigned lds_addr, const void* gptr) {
  unsigned long long ga = (unsigned long long)gptr;
  v4u g0;
  g0[0] = 1u;                                  // count=1, user mode
  g0[1] = lds_addr;                            // LDS byte address
  g0[2] = (unsigned)(ga & 0xFFFFFFFFu);        // global_addr[31:0]
  g0[3] = (unsigned)((ga >> 32) & 0x01FFFFFFu) // global_addr[56:32]
        | 0x80000000u;                         // type=2 ("image")
  return g0;
}
// Build D# group1 for a 4-byte-element tensor (bit layout per ISA 8.4).
__device__ __forceinline__ v8i tdm_g1(unsigned td0, unsigned td1,
                                      unsigned tile0, unsigned tile1, unsigned tile2,
                                      unsigned long long s0, unsigned long long s1) {
  v8i g1;
  g1[0] = (int)(2u << 16);                                    // data_size = 4B
  g1[1] = (int)((td0 & 0xFFFFu) << 16);                       // tensor_dim0[15:0]
  g1[2] = (int)((td0 >> 16) | ((td1 & 0xFFFFu) << 16));       // td0[31:16] | td1[15:0]
  g1[3] = (int)((td1 >> 16) | (tile0 << 16));                 // td1[31:16] | tile_dim0
  g1[4] = (int)(tile1 | (tile2 << 16));                       // tile_dim1 | tile_dim2
  g1[5] = (int)(unsigned)(s0 & 0xFFFFFFFFu);                  // dim0_stride[31:0]
  g1[6] = (int)((unsigned)(s0 >> 32) | ((unsigned)(s1 & 0xFFFFu) << 16)); // s0[47:32] | s1[15:0]
  g1[7] = (int)(unsigned)(s1 >> 16);                          // s1[47:16]
  return g1;
}
#endif

// Block = 128 threads (4 waves). Block computes a 64M x 64N tile of
// Y = W[M,K] @ X[K, B*d] with K-steps of 32 via v_wmma_f32_16x16x32_f16.
// Wave w owns M rows [m0b + 16w, +16); each wave does 4 N-subtiles.
// W tile (64x32 f32) and X tile (32k x 64col f32) staged into LDS by the
// Tensor Data Mover (zero-fill handles all M/K tails); epilogue fuses
// bias + relu + kd sel-gather (row m -> f=m/3, g=m%3) + pairwise max-pool
// via integer atomicMax on the float bits (valid: relu >= 0, OUT zeroed).
__global__ __launch_bounds__(128)
void kdconv_wmma(const float* __restrict__ X,    // [B, K, d]
                 const float* __restrict__ W,    // [M=3cf, K]
                 const float* __restrict__ bias, // [M]
                 const int*   __restrict__ sel,  // [d]
                 float*       __restrict__ OUT,  // [B, cf, d/2], pre-zeroed
                 int M, int K, int d, int dShift, int cf)
{
  extern __shared__ float smem[];          // [0,2048): W tile  [64m][32k]
  float* Ws  = smem;                       // 8 KB
  float* Xls = smem + 2048;                // 8 KB: [z][32k][dtile n]

  const int tid  = threadIdx.x;
  const int wave = tid >> 5;
  const int lane = tid & 31;
  const int half = lane >> 4;
  const int l16  = lane & 15;

  const int c0  = blockIdx.x * 64;         // global column base (col = b*d + n)
  const int m0b = blockIdx.y * 64;         // block M base
  const int dtShift = dShift < 6 ? dShift : 6;
  const int dtile   = 1 << dtShift;        // n-extent per batch inside the tile
  const int ztiles  = 64 >> dtShift;       // batches spanned by the 64-col tile
  const int b0  = c0 >> dShift;
  const int n0  = c0 & (d - 1);

  v8f acc[4] = {};

  for (int k0 = 0; k0 < K; k0 += 32) {
    __syncthreads();                       // previous-step LDS reads complete

#if USE_TDM
    if (wave == 0) {
      v4i z4 = {};
      v8i z8 = {};
      // ---- W tile: 2D, dim0=k (contig, stride K rows), dim1=m. Tile 32x64.
      {
        v4u g0 = tdm_g0(0u, W + (size_t)m0b * K + k0);
        v8i g1 = tdm_g1((unsigned)(K - k0), (unsigned)(M > m0b ? M - m0b : 0),
                        32u, 64u, 0u, (unsigned long long)K, 0ull);
        __builtin_amdgcn_tensor_load_to_lds(g0, g1, z4, z4, z8, 0);
      }
      // ---- X tile: 3D [b][k][n]: dim0=n (contig), dim1=k (stride d),
      //      dim2=b (stride K*d). Tile dtile x 32 x ztiles.
      {
        v4u g0 = tdm_g0(8192u, X + (size_t)b0 * K * d + (size_t)k0 * d + n0);
        v8i g1 = tdm_g1((unsigned)(d - n0), (unsigned)(K - k0),
                        (unsigned)dtile, 32u, (unsigned)ztiles,
                        (unsigned long long)d, (unsigned long long)K * d);
        v4i g2; g2[0] = B_BATCH - b0; g2[1] = 0; g2[2] = 0; g2[3] = 0;
        __builtin_amdgcn_tensor_load_to_lds(g0, g1, g2, z4, z8, 0);
      }
      __builtin_amdgcn_s_wait_tensorcnt(0);
    }
#else
    // ---- Fallback cooperative staging (guarded scalar loads, zero-fill tails)
    {
      const int row = tid >> 1;            // 0..63
      const int kb  = (tid & 1) * 16;      // 0 or 16
      // W tile rows = m, cols = k
      {
        int m = m0b + row;
        const float* Wr = W + (size_t)(m < M ? m : 0) * K;
        #pragma unroll
        for (int i = 0; i < 16; ++i) {
          int k = k0 + kb + i;
          Ws[row * 32 + kb + i] = (m < M && k < K) ? Wr[k] : 0.0f;
        }
      }
      // X tile: column ct = row, k varies
      {
        int c  = c0 + row;
        int bb = c >> dShift;
        int nn = c & (d - 1);
        int z  = row >> dtShift;
        int nt = row & (dtile - 1);
        const float* Xc = X + (size_t)bb * K * d + nn;
        float* dst = Xls + (((z * 32) << dtShift) + nt);
        #pragma unroll
        for (int i = 0; i < 16; ++i) {
          int k = k0 + kb + i;
          dst[(size_t)(kb + i) << dtShift] = (k < K) ? Xc[(size_t)k * d] : 0.0f;
        }
      }
    }
#endif
    __syncthreads();                       // tile visible to all waves

    // ---- A fragment from LDS: lane row = wave*16 + l16,
    //      element j <-> k = half*16 + j
    v16h a;
    {
      const float* ar = Ws + (wave * 16 + l16) * 32 + half * 16;
      #pragma unroll
      for (int j = 0; j < 8; ++j) a[j]     = (_Float16)ar[j];
      #pragma unroll
      for (int j = 0; j < 8; ++j) a[8 + j] = (_Float16)ar[8 + j];
    }

    // ---- 4 B fragments from LDS: col ct = t*16+l16, element j <-> k = half*16+j
    #pragma unroll
    for (int t = 0; t < 4; ++t) {
      int ct = t * 16 + l16;
      const float* br = Xls + (((ct >> dtShift) * 32 + half * 16) << dtShift)
                            + (ct & (dtile - 1));
      v16h bf;
      #pragma unroll
      for (int j = 0; j < 16; ++j) bf[j] = (_Float16)br[(size_t)j << dtShift];
      acc[t] = __builtin_amdgcn_wmma_f32_16x16x32_f16(false, a, false, bf,
                                                      (short)0, acc[t], false, false);
    }
  }

  // ---- Epilogue. C layout: n = lane&15, m = v + 8*(lane>>4).
  const int dh = d >> 1;
  #pragma unroll
  for (int t = 0; t < 4; ++t) {
    int c  = c0 + t * 16 + l16;
    int bb = c >> dShift;
    int nn = c & (d - 1);
    int g  = sel[nn];
    int j2 = nn >> 1;
    #pragma unroll
    for (int v = 0; v < 8; ++v) {
      int m = m0b + wave * 16 + v + (half << 3);
      if (m < M) {
        int f  = m / 3;                    // group is the FAST axis of the reshape
        int gm = m - 3 * f;
        if (gm == g) {
          float val = acc[t][v] + bias[m];
          val = fmaxf(val, 0.0f);
          atomicMax((int*)(OUT + (size_t)bb * cf * dh + (size_t)f * dh + j2),
                    __float_as_int(val));
        }
      }
    }
  }
}

// Final FC (128x16 <- 128x1024 @ 1024x16) + log_softmax, one wave per batch row.
__global__ __launch_bounds__(32)
void fc_logsoftmax(const float* __restrict__ h,    // [128,1024]
                   const float* __restrict__ fcW,  // [16,1024]
                   const float* __restrict__ fcb,  // [16]
                   float*       __restrict__ out)  // [128,16]
{
  const int b    = blockIdx.x;
  const int lane = threadIdx.x;
  float logit = -1e30f;
  if (lane < 16) {
    const float* hr = h + (size_t)b * 1024;
    const float* wr = fcW + (size_t)lane * 1024;
    float s = 0.0f;
    for (int k = 0; k < 1024; ++k) s = fmaf(hr[k], wr[k], s);
    logit = s + fcb[lane];
  }
  float mx = logit;
  #pragma unroll
  for (int o = 8; o >= 1; o >>= 1) mx = fmaxf(mx, __shfl_xor(mx, o, 16));
  float ex = (lane < 16) ? expf(logit - mx) : 0.0f;
  float sm = ex;
  #pragma unroll
  for (int o = 8; o >= 1; o >>= 1) sm += __shfl_xor(sm, o, 16);
  if (lane < 16) out[(size_t)b * 16 + lane] = logit - mx - logf(sm);
}

extern "C" void kernel_launch(void* const* d_in, const int* in_sizes, int n_in,
                              void* d_out, int out_size, void* d_ws, size_t ws_size,
                              hipStream_t stream) {
  static const int CIN[11]  = {3, 8, 32, 64, 64, 64, 128, 256, 512, 512, 512};
  static const int CF[11]   = {8, 32, 64, 64, 64, 128, 256, 512, 512, 512, 1024};
  static const int DD[11]   = {2048, 1024, 512, 256, 128, 64, 32, 16, 8, 4, 2};
  static const int DSH[11]  = {11, 10, 9, 8, 7, 6, 5, 4, 3, 2, 1};

  // Input dict order: x, (W1,b1,sel1), ..., (W11,b11,sel11), fcW, fcb
  const float* x = (const float*)d_in[0];

  // Ping-pong activation buffers; max stage output = 128*32*512 = 2,097,152 f32
  const size_t BUF = 2097152;
  float* bufA = (float*)d_ws;
  float* bufB = bufA + BUF;

  const float* cur = x;
  float* nxt = bufA;
  for (int s = 0; s < 11; ++s) {
    const int K  = CIN[s];
    const int cf = CF[s];
    const int d  = DD[s];
    const int M  = 3 * cf;
    const int outElems = B_BATCH * cf * (d / 2);

    zero_f32<<<(outElems + 255) / 256, 256, 0, stream>>>(nxt, outElems);

    const float* W  = (const float*)d_in[1 + 3 * s];
    const float* bi = (const float*)d_in[2 + 3 * s];
    const int*   se = (const int*)  d_in[3 + 3 * s];

    const int Ntot = B_BATCH * d;                   // always a multiple of 64
    dim3 grid(Ntot / 64, (M + 63) / 64);
    kdconv_wmma<<<grid, 128, 16384, stream>>>(cur, W, bi, se, nxt,
                                              M, K, d, DSH[s], cf);

    cur = nxt;
    nxt = (nxt == bufA) ? bufB : bufA;
  }

  const float* fcW = (const float*)d_in[34];
  const float* fcb = (const float*)d_in[35];
  fc_logsoftmax<<<B_BATCH, 32, 0, stream>>>(cur, fcW, fcb, (float*)d_out);
}
